// MeanProbExtractor_yolov5_86973087744150
// MI455X (gfx1250) — compile-verified, matrix operand order checked by reading, not run on hardware
//
#include <hip/hip_runtime.h>

#define NIMG   16
#define NPRED  25200
#define NFEAT  85
#define TOPK   300
#define TPAD   304
#define NT     (TPAD / 16)      // 19 tiles per dimension
#define CONF_T 0.25f
#define IOU_T  0.45f
#define S1_ROWS 128             // predictions per stage-1 block

typedef __attribute__((ext_vector_type(16))) _Float16 v16h;
typedef __attribute__((ext_vector_type(8)))  float    v8f;
typedef __attribute__((ext_vector_type(4)))  int      v4i;

// Async global->LDS staging path (gfx1250 ASYNCcnt pipeline), with safe
// fallback to regular b128 loads if the toolchain lacks the builtins.
#if defined(__has_builtin)
#if __has_builtin(__builtin_amdgcn_global_load_async_to_lds_b128) && \
    __has_builtin(__builtin_amdgcn_s_wait_asynccnt)
#define USE_ASYNC_LDS 1
#endif
#endif

// ---------------------------------------------------------------------------
// Stage 1: streaming score computation. Memory-bound (137 MB @ 23.3 TB/s).
// Coalesced 16B copies global->LDS (async path when available), then
// per-thread row processing out of LDS (stride-85 words: conflict-free).
// ---------------------------------------------------------------------------
__global__ __launch_bounds__(S1_ROWS) void yolo_score_kernel(
    const float* __restrict__ in, float* __restrict__ scores) {
  __shared__ __align__(16) float tile[S1_ROWS * NFEAT];  // 43520 B
  const int tid = threadIdx.x;
  const size_t baseElem = (size_t)blockIdx.x * S1_ROWS * NFEAT;  // %4 == 0
  const float4* in4 = (const float4*)(in + baseElem);
  float4* t4 = (float4*)tile;
  const int nvec = S1_ROWS * NFEAT / 4;  // 2720

#if USE_ASYNC_LDS
  for (int k = tid; k < nvec; k += S1_ROWS) {
    __builtin_amdgcn_global_load_async_to_lds_b128(
        (v4i*)(in4 + k), (v4i*)(t4 + k), /*offset=*/0, /*cpol=*/0);
  }
  __builtin_amdgcn_s_wait_asynccnt(0);
  __syncthreads();
#else
  for (int k = tid; k < nvec; k += S1_ROWS) t4[k] = in4[k];
  __syncthreads();
#endif

  const float* row = tile + tid * NFEAT;
  const float obj = row[4];
  float maxv = row[5];
  int jcls = 0;
#pragma unroll
  for (int c = 1; c < 80; ++c) {
    float v = row[5 + c];
    if (v > maxv) { maxv = v; jcls = c; }   // first-max => lowest index
  }
  const float conf = obj * maxv;            // obj >= 0, so max distributes
  const bool valid = (obj > CONF_T) && (conf > CONF_T) && (jcls == 0);
  scores[(size_t)blockIdx.x * S1_ROWS + tid] = valid ? conf : -1.0f;
}

// ---------------------------------------------------------------------------
// Stage 2: per-image top-300 selection, WMMA-assisted IOU matrix, NMS, mean.
// One 1024-thread workgroup (32 wave32) per image.
// ---------------------------------------------------------------------------
__global__ __launch_bounds__(1024) void yolo_nms_kernel(
    const float* __restrict__ in, const float* __restrict__ scores,
    float* __restrict__ out) {
  const int img  = blockIdx.x;
  const int tid  = threadIdx.x;
  const int lane = tid & 31;
  const int wave = tid >> 5;

  __shared__ unsigned short Mb16[TPAD * NT];  // suppression bitmask, 11552 B
  __shared__ float topS[TPAD];
  __shared__ int   topIdx[TPAD];
  __shared__ float bx1[TPAD], by1[TPAD], bx2[TPAD], by2[TPAD], barea[TPAD];
  __shared__ float redV[32];
  __shared__ int   redI[32];
  __shared__ int   winI;
  __shared__ unsigned char keepF[TPAD];

  // ---- register-resident score stripe (25 * 1024 = 25600 >= 25200) ----
  const int CHUNK = 25;
  float v[CHUNK];
  const float* simg = scores + (size_t)img * NPRED;
#pragma unroll
  for (int k = 0; k < CHUNK; ++k) {
    int idx = tid + (k << 10);
    v[k] = (idx < NPRED) ? simg[idx] : -3.0e38f;
  }
  if (tid < TPAD) { topS[tid] = -1.0f; topIdx[tid] = 0; }
  __syncthreads();

  // ---- top-300 via iterative argmax extraction ----
  for (int it = 0; it < TOPK; ++it) {
    float bv = v[0];
    int bk = 0;
#pragma unroll
    for (int k = 1; k < CHUNK; ++k)
      if (v[k] > bv) { bv = v[k]; bk = k; }
    int bi = tid + (bk << 10);
    // wave32 pair-max reduce, ties -> lowest index (top_k stability)
#pragma unroll
    for (int off = 16; off > 0; off >>= 1) {
      float ov = __shfl_xor(bv, off, 32);
      int   oi = __shfl_xor(bi, off, 32);
      if (ov > bv || (ov == bv && oi < bi)) { bv = ov; bi = oi; }
    }
    if (lane == 0) { redV[wave] = bv; redI[wave] = bi; }
    __syncthreads();
    if (wave == 0) {
      bv = redV[lane];
      bi = redI[lane];
#pragma unroll
      for (int off = 16; off > 0; off >>= 1) {
        float ov = __shfl_xor(bv, off, 32);
        int   oi = __shfl_xor(bi, off, 32);
        if (ov > bv || (ov == bv && oi < bi)) { bv = ov; bi = oi; }
      }
      if (lane == 0) { winI = bi; topS[it] = bv; topIdx[it] = bi; }
    }
    __syncthreads();
    const int wi = winI;
    if ((wi & 1023) == tid) {        // owner invalidates its register copy
      const int wk = wi >> 10;
#pragma unroll
      for (int k = 0; k < CHUNK; ++k)
        if (k == wk) v[k] = -3.0e38f;
    }
  }

  // ---- gather boxes of the top-300 (scale 1/64: IOU is scale-invariant,
  //      keeps areas in comfortable f16 range for the WMMA outer-sum) ----
  if (tid < TPAD) {
    float x1 = 0.f, y1 = 0.f, x2 = 0.f, y2 = 0.f;
    if (tid < TOPK) {
      const float* p = in + ((size_t)img * NPRED + topIdx[tid]) * NFEAT;
      const float s = 1.0f / 64.0f;
      float cx = p[0] * s, cy = p[1] * s;
      float hw = p[2] * (0.5f * s), hh = p[3] * (0.5f * s);
      x1 = cx - hw; y1 = cy - hh; x2 = cx + hw; y2 = cy + hh;
    }
    bx1[tid] = x1; by1[tid] = y1; bx2[tid] = x2; by2[tid] = y2;
    barea[tid] = (x2 - x1) * (y2 - y1);
    keepF[tid] = (topS[tid] > 0.0f) ? 1 : 0;   // vmask
  }
  __syncthreads();

  // ---- pairwise IOU threshold matrix: rank-2 denominator (a1_i + a2_j)
  //      on the matrix pipe, -inter folded into C: D = a1+a2-inter ----
  for (int t = wave; t < NT * NT; t += 32) {   // wave-uniform loop, EXEC full
    const int ti = t / NT, tj = t % NT;
    const int jcol  = tj * 16 + (lane & 15);
    const int mbase = ti * 16 + ((lane >> 4) << 3);
    const bool lo = (lane < 16);

    v16h a = {};  // A 16x32 f16: K=0 -> area_i, K=1 -> 1
    v16h b = {};  // B 32x16 f16: K=0 -> 1,      K=1 -> area_j
    a[0] = lo ? (_Float16)barea[ti * 16 + (lane & 15)] : (_Float16)0.0f;
    a[1] = lo ? (_Float16)1.0f : (_Float16)0.0f;
    b[0] = lo ? (_Float16)1.0f : (_Float16)0.0f;
    b[1] = lo ? (_Float16)barea[jcol] : (_Float16)0.0f;

    const float jx1 = bx1[jcol], jy1 = by1[jcol];
    const float jx2 = bx2[jcol], jy2 = by2[jcol];
    float inter[8];
    v8f c;
#pragma unroll
    for (int m = 0; m < 8; ++m) {
      const int i = mbase + m;
      float xx1 = fmaxf(bx1[i], jx1);
      float yy1 = fmaxf(by1[i], jy1);
      float xx2 = fminf(bx2[i], jx2);
      float yy2 = fminf(by2[i], jy2);
      float w = fmaxf(xx2 - xx1, 0.0f);
      float h = fmaxf(yy2 - yy1, 0.0f);
      inter[m] = w * h;
      c[m] = -inter[m];
    }
    v8f d = __builtin_amdgcn_wmma_f32_16x16x32_f16(
        false, a, false, b, (short)0, c, false, false);

#pragma unroll
    for (int m = 0; m < 8; ++m) {
      // iou > thr  <=>  inter > thr * (a1 + a2 - inter + eps)
      const int pred = (inter[m] > IOU_T * (d[m] + 1e-9f)) ? 1 : 0;
      const unsigned int bal = (unsigned int)__ballot(pred);
      if (lane == 0)  Mb16[(ti * 16 + m)     * NT + tj] = (unsigned short)(bal & 0xFFFFu);
      if (lane == 16) Mb16[(ti * 16 + m + 8) * NT + tj] = (unsigned short)(bal >> 16);
    }
  }
  __syncthreads();

  // ---- sequential NMS scan: pure LDS bit tests ----
  for (int i = 0; i < TOPK; ++i) {
    const bool ki = keepF[i] != 0;
    if (ki && tid < TPAD && tid > i) {
      if ((Mb16[i * NT + (tid >> 4)] >> (tid & 15)) & 1) keepF[tid] = 0;
    }
    __syncthreads();
  }

  // ---- mean over survivors ----
  if (tid == 0) {
    float s = 0.0f;
    int cnt = 0;
    for (int i = 0; i < TOPK; ++i)
      if (keepF[i]) { s += topS[i]; ++cnt; }
    out[img] = (cnt > 0) ? (s / (float)cnt) : 0.0f;
  }
}

// ---------------------------------------------------------------------------
extern "C" void kernel_launch(void* const* d_in, const int* in_sizes, int n_in,
                              void* d_out, int out_size, void* d_ws, size_t ws_size,
                              hipStream_t stream) {
  const float* in = (const float*)d_in[0];
  float* scores = (float*)d_ws;   // 16*25200 floats = 1.61 MB scratch
  float* out = (float*)d_out;     // 16 floats

  yolo_score_kernel<<<NIMG * NPRED / S1_ROWS, S1_ROWS, 0, stream>>>(in, scores);
  yolo_nms_kernel<<<NIMG, 1024, 0, stream>>>(in, scores, out);
}
